// CLSADecoder_24704651887136
// MI455X (gfx1250) — matrix operand order repaired and checked by reading, not compiled, not run on hardware
//
#include <hip/hip_runtime.h>
#include <math.h>

typedef __attribute__((ext_vector_type(16))) _Float16 v16h;
typedef __attribute__((ext_vector_type(8)))  _Float16 v8h;
typedef __attribute__((ext_vector_type(8)))  float    v8f;

#define B_   64
#define T_   32
#define R_   10
#define C_   10
#define CH_  8
#define D_   800
#define ENC_ 128
#define K2_  1600   // concat(ctx, h) width

__device__ __forceinline__ float sigmoidf_(float x) { return 1.0f / (1.0f + __expf(-x)); }

// ---------------------------------------------------------------------------
// Prep: transpose FC weights [K2, D] (row-major, K-major) -> f16 [D, K2]
// (N-major) so WMMA B-fragments are contiguous 32B runs per lane.
// ---------------------------------------------------------------------------
__global__ void prep_wt_kernel(const float* __restrict__ W, _Float16* __restrict__ Wt) {
    int idx = blockIdx.x * blockDim.x + threadIdx.x;
    if (idx >= K2_ * D_) return;
    int n = idx % D_;
    int k = idx / D_;
    Wt[n * K2_ + k] = (_Float16)W[k * D_ + n];
}

__global__ void copy_f32_kernel(const float* __restrict__ src, float* __restrict__ dst, int n) {
    int i = blockIdx.x * blockDim.x + threadIdx.x;
    if (i < n) dst[i] = src[i];
}

// ---------------------------------------------------------------------------
// Row-shared ConvLSTM cell: 1D conv (k=3, SAME) along cols, gates, state update.
// One thread per (b, r, ch, c). Reads hcur, writes hnew (double-buffered);
// c updated in place (each thread touches only its own element).
// ---------------------------------------------------------------------------
__global__ void conv_cell_kernel(int layer, int s,
                                 const float* __restrict__ x_flat, // layer 0 input [B,T,100]
                                 const float* __restrict__ xin,    // layer 1 input = ref[0][s] [B,D]
                                 const float* __restrict__ hcur,
                                 float* __restrict__ hnew,
                                 float* __restrict__ cstate,
                                 const float* __restrict__ w,
                                 const float* __restrict__ bias) {
    int idx = blockIdx.x * blockDim.x + threadIdx.x;
    if (idx >= B_ * R_ * CH_ * C_) return;
    int c  = idx % C_;  int t = idx / C_;
    int ch = t % CH_;   t /= CH_;
    int r  = t % R_;    int b = t / R_;

    const int cinx   = (layer == 0) ? 1 : CH_;
    const int cinTot = cinx + CH_;

    float acc[4];
#pragma unroll
    for (int q = 0; q < 4; q++) acc[q] = bias[q * CH_ + ch];

    for (int ic = 0; ic < cinTot; ic++) {
#pragma unroll
        for (int k = 0; k < 3; k++) {
            int cc = c + k - 1;
            if (cc < 0 || cc >= C_) continue;
            float z;
            if (ic < cinx) {
                z = (layer == 0) ? x_flat[(b * T_ + s) * (R_ * C_) + r * C_ + cc]
                                 : xin[((b * R_ + r) * CH_ + ic) * C_ + cc];
            } else {
                z = hcur[((b * R_ + r) * CH_ + (ic - cinx)) * C_ + cc];
            }
#pragma unroll
            for (int q = 0; q < 4; q++)
                acc[q] += w[(q * CH_ + ch) * cinTot * 3 + ic * 3 + k] * z;
        }
    }
    float cprev = cstate[idx];
    float i_ = sigmoidf_(acc[0]);
    float f_ = sigmoidf_(acc[1]);
    float g_ = tanhf(acc[2]);
    float o_ = sigmoidf_(acc[3]);
    float c2 = f_ * cprev + i_ * g_;
    cstate[idx] = c2;
    hnew[idx]   = o_ * tanhf(c2);
}

// ---------------------------------------------------------------------------
// Encoder attention: per-batch softmax(h . enc_l) context; emits f16 A = [ctx || h].
// One block per batch row. Batched GEMV with per-batch matrices -> VALU path.
// ---------------------------------------------------------------------------
__global__ void inter_attn_kernel(const float* __restrict__ h,    // [B,D] (hr, post-conv)
                                  const float* __restrict__ enc,  // [B,ENC,D]
                                  _Float16* __restrict__ A) {     // [B, K2]
    __shared__ float sh[D_];
    __shared__ float sw[ENC_];
    int b = blockIdx.x, tid = threadIdx.x;

    for (int d = tid; d < D_; d += blockDim.x) sh[d] = h[b * D_ + d];
    __syncthreads();

    if (tid < ENC_) {
        const float* e = enc + (size_t)(b * ENC_ + tid) * D_;
        float acc = 0.f;
        for (int d = 0; d < D_; d++) acc += sh[d] * e[d];
        sw[tid] = acc;
    }
    __syncthreads();
    if (tid == 0) {
        float m = -1e30f;
        for (int l = 0; l < ENC_; l++) m = fmaxf(m, sw[l]);
        float ssum = 0.f;
        for (int l = 0; l < ENC_; l++) { float e0 = __expf(sw[l] - m); sw[l] = e0; ssum += e0; }
        float inv = 1.f / ssum;
        for (int l = 0; l < ENC_; l++) sw[l] *= inv;
    }
    __syncthreads();
    for (int d = tid; d < D_; d += blockDim.x) {
        float acc = 0.f;
        for (int l = 0; l < ENC_; l++) acc += sw[l] * enc[(size_t)(b * ENC_ + l) * D_ + d];
        A[(size_t)b * K2_ + d]      = (_Float16)acc;
        A[(size_t)b * K2_ + D_ + d] = (_Float16)sh[d];
    }
}

// ---------------------------------------------------------------------------
// Self attention over history (t < s): keys = pre, values = ref. Emits A = [ctx || h].
// ---------------------------------------------------------------------------
__global__ void self_attn_kernel(int l, int s,
                                 const float* __restrict__ pre,  // [L,T,B,D]
                                 const float* __restrict__ ref,  // [L,T,B,D]
                                 _Float16* __restrict__ A) {
    __shared__ float sh[D_];
    __shared__ float sw[T_];
    int b = blockIdx.x, tid = threadIdx.x;

    const float* hv = pre + ((size_t)(l * T_ + s) * B_ + b) * D_;
    for (int d = tid; d < D_; d += blockDim.x) sh[d] = hv[d];
    __syncthreads();

    if (tid < s) {
        const float* kk = pre + ((size_t)(l * T_ + tid) * B_ + b) * D_;
        float acc = 0.f;
        for (int d = 0; d < D_; d++) acc += sh[d] * kk[d];
        sw[tid] = acc;
    }
    __syncthreads();
    if (tid == 0) {
        float m = -1e30f;
        for (int t = 0; t < s; t++) m = fmaxf(m, sw[t]);
        float ssum = 0.f;
        for (int t = 0; t < s; t++) { float e0 = __expf(sw[t] - m); sw[t] = e0; ssum += e0; }
        float inv = 1.f / ssum;
        for (int t = 0; t < s; t++) sw[t] *= inv;
    }
    __syncthreads();
    for (int d = tid; d < D_; d += blockDim.x) {
        float acc = 0.f;
        for (int t = 0; t < s; t++)
            acc += sw[t] * ref[((size_t)(l * T_ + t) * B_ + b) * D_ + d];
        A[(size_t)b * K2_ + d]      = (_Float16)acc;
        A[(size_t)b * K2_ + D_ + d] = (_Float16)sh[d];
    }
}

// ---------------------------------------------------------------------------
// WMMA GEMM: out[64,800] = tanh(A[64,1600](f16) @ W(f16, pre-transposed [800,1600]) + bias)
// 200 tiles (4 M x 50 N) = 25 blocks x 8 wave32 waves; K loop = 50 x
// v_wmma_f32_16x16x32_f16 with f32 accumulation. No divergence (EXEC all-1s).
// ---------------------------------------------------------------------------
__global__ void fc_wmma_tanh_kernel(const _Float16* __restrict__ A,   // [64, K2]
                                    const _Float16* __restrict__ Wt,  // [D_, K2]
                                    const float* __restrict__ bias,   // [D_]
                                    float* __restrict__ out) {        // [64, D_]
    int wave = threadIdx.x >> 5;
    int lane = threadIdx.x & 31;
    int tile = blockIdx.x * 8 + wave;   // 0..199
    int mt   = tile & 3;                // M tile 0..3
    int nt   = tile >> 2;               // N tile 0..49
    int half = lane >> 4;
    int lr   = lane & 15;

    const _Float16* aptr = A  + (size_t)(mt * 16 + lr) * K2_;
    const _Float16* bptr = Wt + (size_t)(nt * 16 + lr) * K2_;

    v8f acc = {};
    for (int k0 = 0; k0 < K2_; k0 += 32) {
        // A-fragment, contiguous k order used here: half 0 lanes own K {0..15}, half 1 own K {16..31}
        v8h alo = *(const v8h*)(aptr + k0 + half * 16);
        v8h ahi = *(const v8h*)(aptr + k0 + half * 16 + 8);
        v16h a;
#pragma unroll
        for (int i = 0; i < 8; i++) { a[i] = alo[i]; a[8 + i] = ahi[i]; }
        // B-fragment: lane = column, halves split K {0..15}/{16..31}; contiguous in Wt
        v16h bm = *(const v16h*)(bptr + k0 + half * 16);
        acc = __builtin_amdgcn_wmma_f32_16x16x32_f16(false, a, false, bm,
                                                     (short)0, acc, false, false);
    }
#pragma unroll
    for (int j = 0; j < 8; j++) {
        int row = mt * 16 + j + half * 8;   // C/D layout: vgpr j -> M=j (+8 for high half)
        int col = nt * 16 + lr;
        out[(size_t)row * D_ + col] = tanhf(acc[j] + bias[col]);
    }
}

// ---------------------------------------------------------------------------
// MLP head: 800->200 relu -> 50 relu -> 3. One block per batch row (LDS staged).
// ---------------------------------------------------------------------------
__global__ void head_kernel(int s, const float* __restrict__ hv, // ref[L-1][s]: [B,D]
                            const float* __restrict__ w1, const float* __restrict__ b1,
                            const float* __restrict__ w2, const float* __restrict__ b2,
                            const float* __restrict__ w3, const float* __restrict__ b3,
                            float* __restrict__ out) {           // [B,T,3]
    __shared__ float v0[D_];
    __shared__ float v1[200];
    __shared__ float v2[50];
    int b = blockIdx.x, tid = threadIdx.x;
    for (int d = tid; d < D_; d += blockDim.x) v0[d] = hv[(size_t)b * D_ + d];
    __syncthreads();
    if (tid < 200) {
        float acc = b1[tid];
        for (int k = 0; k < D_; k++) acc += v0[k] * w1[k * 200 + tid];
        v1[tid] = fmaxf(acc, 0.f);
    }
    __syncthreads();
    if (tid < 50) {
        float acc = b2[tid];
        for (int k = 0; k < 200; k++) acc += v1[k] * w2[k * 50 + tid];
        v2[tid] = fmaxf(acc, 0.f);
    }
    __syncthreads();
    if (tid < 3) {
        float acc = b3[tid];
        for (int k = 0; k < 50; k++) acc += v2[k] * w3[k * 3 + tid];
        out[(size_t)(b * T_ + s) * 3 + tid] = acc;
    }
}

// ---------------------------------------------------------------------------
extern "C" void kernel_launch(void* const* d_in, const int* in_sizes, int n_in,
                              void* d_out, int out_size, void* d_ws, size_t ws_size,
                              hipStream_t stream) {
    (void)in_sizes; (void)n_in; (void)out_size; (void)ws_size;

    const float* x_flat = (const float*)d_in[0];
    const float* enc    = (const float*)d_in[1];
    const float* init_h = (const float*)d_in[2];
    const float* init_c = (const float*)d_in[3];
    const float* conv_w[2] = { (const float*)d_in[4],  (const float*)d_in[6]  };
    const float* conv_b[2] = { (const float*)d_in[5],  (const float*)d_in[7]  };
    const float* inter_w[2]= { (const float*)d_in[8],  (const float*)d_in[10] };
    const float* inter_b[2]= { (const float*)d_in[9],  (const float*)d_in[11] };
    const float* self_w[2] = { (const float*)d_in[12], (const float*)d_in[14] };
    const float* self_b[2] = { (const float*)d_in[13], (const float*)d_in[15] };
    const float* head_w1 = (const float*)d_in[16];
    const float* head_b1 = (const float*)d_in[17];
    const float* head_w2 = (const float*)d_in[18];
    const float* head_b2 = (const float*)d_in[19];
    const float* head_w3 = (const float*)d_in[20];
    const float* head_b3 = (const float*)d_in[21];
    float* out = (float*)d_out;

    // -------- workspace layout (256B aligned sub-allocations) --------
    char* wsb = (char*)d_ws;
    size_t off = 0;
    auto alloc = [&](size_t bytes) -> char* {
        char* p = wsb + off;
        off = (off + bytes + 255) & ~(size_t)255;
        return p;
    };
    _Float16* Wt[4];                              // [0]=inter0 [1]=inter1 [2]=self0 [3]=self1
    for (int i = 0; i < 4; i++) Wt[i] = (_Float16*)alloc((size_t)D_ * K2_ * sizeof(_Float16));
    _Float16* Abuf = (_Float16*)alloc((size_t)B_ * K2_ * sizeof(_Float16));
    float* hbuf[2][2];
    float* cbuf[2];
    for (int l = 0; l < 2; l++) {
        hbuf[l][0] = (float*)alloc((size_t)B_ * D_ * sizeof(float));
        hbuf[l][1] = (float*)alloc((size_t)B_ * D_ * sizeof(float));
        cbuf[l]    = (float*)alloc((size_t)B_ * D_ * sizeof(float));
    }
    float* pre = (float*)alloc((size_t)2 * T_ * B_ * D_ * sizeof(float));
    float* ref = (float*)alloc((size_t)2 * T_ * B_ * D_ * sizeof(float));

    // -------- one-time prep per launch (deterministic) --------
    {
        int nw = K2_ * D_, gb = (nw + 255) / 256;
        prep_wt_kernel<<<gb, 256, 0, stream>>>(inter_w[0], Wt[0]);
        prep_wt_kernel<<<gb, 256, 0, stream>>>(inter_w[1], Wt[1]);
        prep_wt_kernel<<<gb, 256, 0, stream>>>(self_w[0],  Wt[2]);
        prep_wt_kernel<<<gb, 256, 0, stream>>>(self_w[1],  Wt[3]);
        int ns = B_ * D_, gs = (ns + 255) / 256;
        for (int l = 0; l < 2; l++) {
            copy_f32_kernel<<<gs, 256, 0, stream>>>(init_h + (size_t)l * ns, hbuf[l][0], ns);
            copy_f32_kernel<<<gs, 256, 0, stream>>>(init_c + (size_t)l * ns, cbuf[l], ns);
        }
    }

    const int convBlocks = (B_ * R_ * CH_ * C_ + 255) / 256;
    const int copyBlocks = (B_ * D_ + 255) / 256;

    // -------- sequential scan over T steps --------
    for (int s = 0; s < T_; s++) {
        for (int l = 0; l < 2; l++) {
            const float* hcur = hbuf[l][s & 1];
            float*       hnew = hbuf[l][(s + 1) & 1];
            const float* xin1 = ref + ((size_t)(0 * T_ + s) * B_) * D_;  // layer-1 input

            conv_cell_kernel<<<convBlocks, 256, 0, stream>>>(
                l, s, x_flat, xin1, hcur, hnew, cbuf[l], conv_w[l], conv_b[l]);

            inter_attn_kernel<<<B_, 256, 0, stream>>>(hnew, enc, Abuf);

            float* preLS = pre + ((size_t)(l * T_ + s) * B_) * D_;
            fc_wmma_tanh_kernel<<<25, 256, 0, stream>>>(Abuf, Wt[l], inter_b[l], preLS);

            float* refLS = ref + ((size_t)(l * T_ + s) * B_) * D_;
            if (s == 0) {
                copy_f32_kernel<<<copyBlocks, 256, 0, stream>>>(preLS, refLS, B_ * D_);
            } else {
                self_attn_kernel<<<B_, 256, 0, stream>>>(l, s, pre, ref, Abuf);
                fc_wmma_tanh_kernel<<<25, 256, 0, stream>>>(Abuf, Wt[2 + l], self_b[l], refLS);
            }
        }
        head_kernel<<<B_, 256, 0, stream>>>(
            s, ref + ((size_t)(1 * T_ + s) * B_) * D_,
            head_w1, head_b1, head_w2, head_b2, head_w3, head_b3, out);
    }
}
